// ResidualVectorQuantizer_45698452029652
// MI455X (gfx1250) — matrix operand order repaired and checked by reading, not compiled, not run on hardware
//
#include <hip/hip_runtime.h>
#include <hip/hip_bf16.h>

// ---------------- problem constants ----------------
constexpr int kB   = 16;     // batch
constexpr int kD   = 256;    // codebook dim
constexpr int kT   = 2048;   // time steps
constexpr int kK   = 1024;   // codebook size
constexpr int kNCB = 8;      // number of codebooks
constexpr int kBT  = kB * kT;                 // 32768 rows
constexpr long long kQN = (long long)kB * kD * kT;   // 8388608 quantized elems
constexpr long long kIN = (long long)kB * kNCB * kT; // 262144 index elems

typedef __attribute__((ext_vector_type(16))) __bf16 v16bf;
typedef __attribute__((ext_vector_type(8)))  float  v8f;
typedef __attribute__((ext_vector_type(4)))  float  vf4;

// ---------------- prep kernels ----------------

// fp32 -> bf16 codebook copy (same [ncb][K][D] layout; D contiguous per row,
// which is exactly what the B-fragment per-lane load wants).
__global__ __launch_bounds__(256) void cvt_cb_bf16(const float* __restrict__ cb,
                                                   __bf16* __restrict__ cb16) {
  size_t i = (size_t)blockIdx.x * 256 + threadIdx.x;  // kNCB*kK*kD total
  cb16[i] = (__bf16)cb[i];
}

// c2[i*K + k] = sum_d cb[i][k][d]^2 ; thread 0 also zeroes the loss accumulator.
__global__ __launch_bounds__(256) void calc_c2(const float* __restrict__ cb,
                                               float* __restrict__ c2,
                                               float* __restrict__ loss_acc) {
  int i = blockIdx.x * 256 + threadIdx.x;  // kNCB*kK = 8192 rows
  if (i == 0) *loss_acc = 0.0f;
  const float* p = cb + (size_t)i * kD;
  float s = 0.0f;
#pragma unroll 8
  for (int d = 0; d < kD; ++d) { float v = p[d]; s += v * v; }
  c2[i] = s;
}

// x [B][D][T] -> resid [B][T][D], LDS 32x32 tile transpose.
__global__ __launch_bounds__(256) void transpose_x(const float* __restrict__ x,
                                                   float* __restrict__ resid) {
  __shared__ float tile[32][33];
  int b  = blockIdx.z;
  int t0 = blockIdx.x * 32;
  int d0 = blockIdx.y * 32;
  int tx = threadIdx.x, ty = threadIdx.y;  // 32 x 8
#pragma unroll
  for (int j = 0; j < 32; j += 8)
    tile[ty + j][tx] = x[((size_t)b * kD + (d0 + ty + j)) * kT + t0 + tx];
  __syncthreads();
#pragma unroll
  for (int j = 0; j < 32; j += 8)
    resid[((size_t)b * kT + (t0 + ty + j)) * kD + d0 + tx] = tile[tx][ty + j];
}

// ---------------- stage kernel 1: WMMA distances + argmin ----------------
// A-fragment (16x32 bf16) element packing (contiguous k, matching B):
//   lanes 0-15 : M=lane,    elems 0..15 = K base+0..15 (base = ks*32)
//   lanes16-31 : M=lane-16, base shifted by +16
// p points at (row, ks*32 + half*16); elems 0..15 <- p[0..15].
static __device__ __forceinline__ v16bf pack_a(const float* __restrict__ p) {
  vf4 f0 = *(const vf4*)(p);
  vf4 f1 = *(const vf4*)(p + 4);
  vf4 f2 = *(const vf4*)(p + 8);
  vf4 f3 = *(const vf4*)(p + 12);
  v16bf r;
  r[0]  = (__bf16)f0.x; r[1]  = (__bf16)f0.y; r[2]  = (__bf16)f0.z; r[3]  = (__bf16)f0.w;
  r[4]  = (__bf16)f1.x; r[5]  = (__bf16)f1.y; r[6]  = (__bf16)f1.z; r[7]  = (__bf16)f1.w;
  r[8]  = (__bf16)f2.x; r[9]  = (__bf16)f2.y; r[10] = (__bf16)f2.z; r[11] = (__bf16)f2.w;
  r[12] = (__bf16)f3.x; r[13] = (__bf16)f3.y; r[14] = (__bf16)f3.z; r[15] = (__bf16)f3.w;
  return r;
}

__global__ __launch_bounds__(128) void rvq_dist_argmin(
    const float*  __restrict__ resid,   // [BT][D] fp32
    const __bf16* __restrict__ cb16,    // stage slice [K][D] bf16
    const float*  __restrict__ c2,      // stage slice [K]
    int*          __restrict__ idx_out, // full index region [B][NCB][T]
    int stage) {
  const int wave = threadIdx.x >> 5;   // 0..3
  const int lane = threadIdx.x & 31;
  const int half = lane >> 4;
  const int l16  = lane & 15;
  const int bt0  = blockIdx.x * 16;    // 16-row M tile

  // A fragments for the whole K=256 reduction (8 steps of K=32), fp32->bf16 in-register.
  v16bf a[8];
  {
    const float* rp = resid + (size_t)(bt0 + l16) * kD;
#pragma unroll
    for (int ks = 0; ks < 8; ++ks)
      a[ks] = pack_a(rp + ks * 32 + half * 16);
  }

  // running argmin: best[v] belongs to row M = v + half*8, column set {wave's N's}.
  // Within a lane n strictly increases over nt, so a strict '<' keeps the
  // first (lowest-index) minimum -- no tie-break VALU needed in the hot loop.
  float best[8]; int bidx[8];
#pragma unroll
  for (int v = 0; v < 8; ++v) { best[v] = 3.0e38f; bidx[v] = 0; }

  const int ncol0 = wave * 256;  // each wave covers 256 codewords = 16 N-tiles
#pragma unroll
  for (int nt = 0; nt < 16; ++nt) {
    const int n = ncol0 + nt * 16 + l16;  // this lane's N (same for all 8 C rows)
    // B-fragment: lane = N, elems = K; K range = ks*32 + half*16 .. +15 (contiguous in d)
    const __bf16* bp = cb16 + (size_t)n * kD + half * 16;
    if (nt + 1 < 16)  // speculative prefetch of next tile's row (L2 temporal)
      __builtin_prefetch(cb16 + (size_t)(n + 16) * kD, 0, 1);

    // Preload ALL 8 B fragments first so the 16 b128 loads issue as one clause
    // and the WMMA chain runs on counted waits instead of load->mma lockstep.
    v16bf bvs[8];
#pragma unroll
    for (int ks = 0; ks < 8; ++ks)
      bvs[ks] = *reinterpret_cast<const v16bf*>(bp + ks * 32);

    v8f c = {};
#pragma unroll
    for (int ks = 0; ks < 8; ++ks)
      c = __builtin_amdgcn_wmma_f32_16x16x32_bf16(false, a[ks], false, bvs[ks],
                                                  (short)0, c, false, false);

    const float cc2 = c2[n];
#pragma unroll
    for (int v = 0; v < 8; ++v) {
      // dist = ||r||^2 - 2 r.c + ||c||^2 ; ||r||^2 is row-constant -> drop for argmin
      float dval = __builtin_fmaf(-2.0f, c[v], cc2);
      if (dval < best[v]) { best[v] = dval; bidx[v] = n; }
    }
  }

  // cross-lane / cross-wave argmin reduce: 64 candidates per row
  __shared__ float sval[16][64];
  __shared__ int   sidx[16][64];
  const int slot = wave * 16 + l16;
#pragma unroll
  for (int v = 0; v < 8; ++v) {
    int m = v + half * 8;
    sval[m][slot] = best[v];
    sidx[m][slot] = bidx[v];
  }
  __syncthreads();

  if (threadIdx.x < 16) {
    int m = threadIdx.x;
    float bv = sval[m][0]; int bi = sidx[m][0];
    for (int s = 1; s < 64; ++s) {
      float v = sval[m][s]; int id = sidx[m][s];
      if (v < bv || (v == bv && id < bi)) { bv = v; bi = id; }
    }
    int row = bt0 + m;
    int b = row >> 11;          // / kT
    int t = row & (kT - 1);
    idx_out[((size_t)b * kNCB + stage) * kT + t] = bi;
  }
}

// ---------------- stage kernel 2: residual update + commitment loss ----------------
__global__ __launch_bounds__(256) void rvq_update(
    float*       __restrict__ resid,    // [BT][D]
    const float* __restrict__ cb,       // stage slice [K][D] fp32
    const int*   __restrict__ idx_out,  // full index region
    int stage,
    float*       __restrict__ loss_acc) {
  int bt = blockIdx.x;
  int d  = threadIdx.x;
  int b = bt >> 11, t = bt & (kT - 1);
  int id = idx_out[((size_t)b * kNCB + stage) * kT + t];  // scalar per block
  float q = cb[(size_t)id * kD + d];
  size_t o = (size_t)bt * kD + d;
  float r = resid[o] - q;
  resid[o] = r;
  float diff = r - q;  // mse(new_residual, quant_i)

  __shared__ float red[256];
  red[d] = diff * diff;
  __syncthreads();
#pragma unroll
  for (int s = 128; s > 0; s >>= 1) {
    if (d < s) red[d] += red[d + s];
    __syncthreads();
  }
  if (d == 0)
    __hip_atomic_fetch_add(loss_acc, red[0], __ATOMIC_RELAXED,
                           __HIP_MEMORY_SCOPE_AGENT);
}

// ---------------- finalize: quantized = x - residual (transpose back) + loss ----------------
__global__ __launch_bounds__(256) void finalize(const float* __restrict__ x,
                                                const float* __restrict__ resid,
                                                float* __restrict__ out,
                                                const float* __restrict__ loss_acc) {
  __shared__ float tile[32][33];
  int b  = blockIdx.z;
  int t0 = blockIdx.x * 32;
  int d0 = blockIdx.y * 32;
  int tx = threadIdx.x, ty = threadIdx.y;  // 32 x 8
#pragma unroll
  for (int j = 0; j < 32; j += 8)
    tile[ty + j][tx] = resid[((size_t)b * kT + (t0 + ty + j)) * kD + d0 + tx];
  __syncthreads();
#pragma unroll
  for (int j = 0; j < 32; j += 8) {
    size_t o = ((size_t)b * kD + (d0 + ty + j)) * kT + t0 + tx;
    out[o] = x[o] - tile[tx][ty + j];
  }
  if (blockIdx.x == 0 && blockIdx.y == 0 && blockIdx.z == 0 && tx == 0 && ty == 0)
    out[kQN + kIN] = *loss_acc * (1.0f / (float)kQN);  // sum of 8 per-stage means
}

// ---------------- host launch ----------------
extern "C" void kernel_launch(void* const* d_in, const int* in_sizes, int n_in,
                              void* d_out, int out_size, void* d_ws, size_t ws_size,
                              hipStream_t stream) {
  const float* x  = (const float*)d_in[0];  // [16,256,2048]
  const float* cb = (const float*)d_in[1];  // [8,1024,256]
  float* out = (float*)d_out;

  // workspace layout
  char* ws = (char*)d_ws;
  float*  resid = (float*)ws;                                   // 33,554,432 B
  size_t  off   = (size_t)kBT * kD * sizeof(float);
  __bf16* cb16  = (__bf16*)(ws + off);                          // 4,194,304 B
  off += (size_t)kNCB * kK * kD * sizeof(__bf16);
  float*  c2    = (float*)(ws + off);                           // 32,768 B
  off += (size_t)kNCB * kK * sizeof(float);
  float*  loss  = (float*)(ws + off);                           // 4 B

  int* idx_out = (int*)((float*)d_out + kQN);  // indices region of d_out

  // prep
  cvt_cb_bf16<<<(kNCB * kK * kD) / 256, 256, 0, stream>>>(cb, cb16);
  calc_c2<<<(kNCB * kK) / 256, 256, 0, stream>>>(cb, c2, loss);
  {
    dim3 g(kT / 32, kD / 32, kB), blk(32, 8);
    transpose_x<<<g, blk, 0, stream>>>(x, resid);
  }

  // 8 sequential RVQ stages
  for (int i = 0; i < kNCB; ++i) {
    const __bf16* cb16_i = cb16 + (size_t)i * kK * kD;
    const float*  cb_i   = cb   + (size_t)i * kK * kD;
    const float*  c2_i   = c2   + (size_t)i * kK;
    rvq_dist_argmin<<<kBT / 16, 128, 0, stream>>>(resid, cb16_i, c2_i, idx_out, i);
    rvq_update<<<kBT, 256, 0, stream>>>(resid, cb_i, idx_out, i, loss);
  }

  // outputs
  {
    dim3 g(kT / 32, kD / 32, kB), blk(32, 8);
    finalize<<<g, blk, 0, stream>>>(x, resid, out, loss);
  }
}
